// MultiRingPrior_9878424781490
// MI455X (gfx1250) — compile-verified, matrix operand order checked by reading, not run on hardware
//
#include <hip/hip_runtime.h>
#include <hip/hip_bf16.h>
#include <math.h>

// ---------------------------------------------------------------------------
// MultiRingPrior for gfx1250 (MI455X, wave32, WMMA)
//
// Stage 1 (1 wave): MLP head via v_wmma_f32_16x16x32_f16 (padded 16x16 tiles,
//                   row 0 = the real 1x192 activation). Writes 14 params to ws.
// Stage 2 (grid):   8.4M samples, Philox-style counter RNG + Box-Muller,
//                   non-temporal packed stores (write-once 100MB, keep L2 clean).
// ---------------------------------------------------------------------------

typedef __attribute__((ext_vector_type(16))) _Float16 v16h;
typedef __attribute__((ext_vector_type(8)))  float    v8f;
typedef __attribute__((ext_vector_type(2)))  float    f32x2;

#define KCOMP   3
#define NCOMP   5
#define CONDD   64
#define HD      128
#define DIN     192        // CONDD + HD
#define HID     128
#define TWO_PI  6.28318530717958647692f

// ---- WMMA fragment builders -------------------------------------------------
// A (16x32 f16, MxK): lane<16 holds row M=lane : K = {k0..k0+7, k0+16..k0+23}
//                     lane>=16 holds row M=lane-16: K = {k0+8..k0+15, k0+24..k0+31}
// Only row 0 is non-zero (our activation vector), everything else stays 0.
__device__ inline v16h make_a_frag(const _Float16* __restrict__ src, int k0, int lane) {
    v16h a = {};
    if (lane == 0) {
#pragma unroll
        for (int j = 0; j < 8; ++j) { a[j] = src[k0 + j];     a[8 + j] = src[k0 + 16 + j]; }
    } else if (lane == 16) {
#pragma unroll
        for (int j = 0; j < 8; ++j) { a[j] = src[k0 + 8 + j]; a[8 + j] = src[k0 + 24 + j]; }
    }
    return a;
}

// B (32x16 f16, KxN), W row-major f32 [K x ncols]:
//   lane<16: col N=lane,    K = {k0..k0+7, k0+16..k0+23}
//   lane>=16: col N=lane-16, K = {k0+8..k0+15, k0+24..k0+31}
__device__ inline v16h make_b_frag(const float* __restrict__ W, int ncols,
                                   int k0, int n0, int lane) {
    const int n     = n0 + (lane & 15);
    const int kbase = k0 + ((lane & 16) ? 8 : 0);
    v16h b = {};
#pragma unroll
    for (int j = 0; j < 8; ++j) {
        b[j]     = (_Float16)W[(kbase + j)      * ncols + n];
        b[8 + j] = (_Float16)W[(kbase + 16 + j) * ncols + n];
    }
    return b;
}

__device__ inline float silu(float z) { return z / (1.0f + __expf(-z)); }

// ---- Stage 1: MLP head, single wave ----------------------------------------
__global__ __launch_bounds__(32)
void mrp_mlp_kernel(const float* __restrict__ c,   const float* __restrict__ h,
                    const float* __restrict__ W1,  const float* __restrict__ b1,
                    const float* __restrict__ W2,  const float* __restrict__ b2,
                    const float* __restrict__ W3,  const float* __restrict__ b3,
                    const float* __restrict__ Wmu, const float* __restrict__ bmu,
                    const float* __restrict__ WR,  const float* __restrict__ bR,
                    float* __restrict__ P /* ws: [0..4]=cdf [5..10]=mus [11..13]=Rs */) {
    __shared__ float     sxf[DIN];
    __shared__ _Float16  sxh[DIN];
    __shared__ _Float16  sz1h[HID];
    __shared__ float     sz2f[HID];
    __shared__ float     heads[16];   // 0..4 logits, 5..10 mus, 11..13 Rs

    const int lane = threadIdx.x;

    for (int i = lane; i < DIN; i += 32) {
        float v = (i < CONDD) ? c[i] : h[i - CONDD];
        sxf[i] = v; sxh[i] = (_Float16)v;
    }
    __syncthreads();

    // ---- layer 1: z1 = silu(x @ W1 + b1), K=192 -> 6 k-steps, 8 n-tiles ----
    for (int n0 = 0; n0 < HID; n0 += 16) {
        v8f acc = {};
#pragma unroll
        for (int k0 = 0; k0 < DIN; k0 += 32) {
            v16h a = make_a_frag(sxh, k0, lane);
            v16h b = make_b_frag(W1, HID, k0, n0, lane);
            acc = __builtin_amdgcn_wmma_f32_16x16x32_f16(false, a, false, b,
                                                         (short)0, acc, false, false);
        }
        if (lane < 16) {                       // D row M=0 lives in acc[0], lanes 0..15
            float s = silu(acc[0] + b1[n0 + lane]);
            sz1h[n0 + lane] = (_Float16)s;
        }
    }
    __syncthreads();

    // ---- layer 2: z2 = silu(z1 @ W2 + b2), K=128 -> 4 k-steps, 8 n-tiles ---
    for (int n0 = 0; n0 < HID; n0 += 16) {
        v8f acc = {};
#pragma unroll
        for (int k0 = 0; k0 < HID; k0 += 32) {
            v16h a = make_a_frag(sz1h, k0, lane);
            v16h b = make_b_frag(W2, HID, k0, n0, lane);
            acc = __builtin_amdgcn_wmma_f32_16x16x32_f16(false, a, false, b,
                                                         (short)0, acc, false, false);
        }
        if (lane < 16) sz2f[n0 + lane] = silu(acc[0] + b2[n0 + lane]);
    }
    __syncthreads();

    // ---- small heads: 5 logits, 6 mus, 3 radii (lane-parallel dots) --------
    if (lane < NCOMP) {
        float acc = b3[lane];
        for (int k = 0; k < HID; ++k) acc += sz2f[k] * W3[k * NCOMP + lane];
        heads[lane] = acc;
    }
    if (lane < 2 * KCOMP) {
        float acc = bmu[lane];
        for (int k = 0; k < DIN; ++k) acc += sxf[k] * Wmu[k * (2 * KCOMP) + lane];
        heads[5 + lane] = acc;
    }
    if (lane < KCOMP) {
        float acc = bR[lane];
        for (int k = 0; k < DIN; ++k) acc += sxf[k] * WR[k * KCOMP + lane];
        heads[11 + lane] = acc;
    }
    __syncthreads();

    // ---- softmax + background cap + CDF (lane 0) ---------------------------
    if (lane == 0) {
        float m = heads[0];
        for (int i = 1; i < NCOMP; ++i) m = fmaxf(m, heads[i]);
        float w[NCOMP], s = 0.0f;
        for (int i = 0; i < NCOMP; ++i) { w[i] = __expf(heads[i] - m); s += w[i]; }
        for (int i = 0; i < NCOMP; ++i) w[i] /= s;
        float wbg = fminf(w[NCOMP - 1], 0.15f);
        float rest = 0.0f;
        for (int i = 0; i < NCOMP - 1; ++i) rest += w[i];
        rest = fmaxf(rest, 1e-8f);
        float scale = (1.0f - wbg) / rest;
        for (int i = 0; i < NCOMP - 1; ++i) w[i] = w[i] * scale + 1e-8f;
        w[NCOMP - 1] = wbg + 1e-8f;
        float tot = 0.0f;
        for (int i = 0; i < NCOMP; ++i) tot += w[i];
        float cdf = 0.0f;
        for (int i = 0; i < NCOMP; ++i) { cdf += w[i] / tot; P[i] = cdf; }
        for (int i = 0; i < 2 * KCOMP; ++i) P[5 + i]  = heads[5 + i];
        for (int i = 0; i < KCOMP; ++i)     P[11 + i] = heads[11 + i];
    }
}

// ---- Stage 2: sampler -------------------------------------------------------
struct U4 { unsigned x, y, z, w; };

__device__ inline U4 philox4(unsigned c0, unsigned c1, unsigned c2, unsigned c3,
                             unsigned k0, unsigned k1) {
    const unsigned M0 = 0xD2511F53u, M1 = 0xCD9E8D57u;
#pragma unroll
    for (int r = 0; r < 7; ++r) {
        unsigned h0 = __umulhi(M0, c0), l0 = M0 * c0;
        unsigned h1 = __umulhi(M1, c2), l1 = M1 * c2;
        unsigned n0 = h1 ^ c1 ^ k0, n1 = l1;
        unsigned n2 = h0 ^ c3 ^ k1, n3 = l0;
        c0 = n0; c1 = n1; c2 = n2; c3 = n3;
        k0 += 0x9E3779B9u; k1 += 0xBB67AE85u;
    }
    return {c0, c1, c2, c3};
}

__device__ inline float u01(unsigned r) { return (float)(r >> 8) * 0x1p-24f; }

__global__ __launch_bounds__(256)
void mrp_sample_kernel(const float* __restrict__ P, float* __restrict__ out_xy,
                       int* __restrict__ out_idx, long long N) {
    // uniform loads -> SGPRs, live for the whole loop
    const float cdf0 = P[0], cdf1 = P[1], cdf2 = P[2], cdf3 = P[3];
    const float mu0x = P[5], mu0y = P[6], mu1x = P[7], mu1y = P[8], mu2x = P[9], mu2y = P[10];
    const float R0 = P[11], R1 = P[12], R2 = P[13];

    const long long stride = (long long)gridDim.x * blockDim.x;
    for (long long i = (long long)blockIdx.x * blockDim.x + threadIdx.x; i < N; i += stride) {
        const unsigned lo = (unsigned)(i & 0xffffffffu), hi = (unsigned)(i >> 32);
        U4 ra = philox4(lo, hi, 0u, 0x0badf00du, 42u, 0x2545F491u);
        U4 rb = philox4(lo, hi, 1u, 0x0badf00du, 42u, 0x2545F491u);

        const float u0 = u01(ra.x), u1 = u01(ra.y), u2 = u01(ra.z), u3 = u01(ra.w);
        const float u4 = u01(rb.x), u5 = u01(rb.y);

        // categorical via inverse CDF
        int idx = (u0 >= cdf0) + (u0 >= cdf1) + (u0 >= cdf2) + (u0 >= cdf3);
        int jr  = idx < (KCOMP - 1) ? idx : (KCOMP - 1);

        float Rj  = (jr == 0) ? R0   : ((jr == 1) ? R1   : R2);
        float mjx = (jr == 0) ? mu0x : ((jr == 1) ? mu1x : mu2x);
        float mjy = (jr == 0) ? mu0y : ((jr == 1) ? mu1y : mu2y);

        // ring point
        float st, ct; __sincosf(TWO_PI * u1, &st, &ct);
        float nr  = __fsqrt_rn(-2.0f * __logf(fmaxf(u2, 1e-9f))) * __cosf(TWO_PI * u3);
        float rad = Rj + 0.02f * nr;
        float ringx = fmaf(rad, ct, mjx), ringy = fmaf(rad, st, mjy);

        // shared gaussian pair for center / background
        float re = __fsqrt_rn(-2.0f * __logf(fmaxf(u4, 1e-9f)));
        float se, ce; __sincosf(TWO_PI * u5, &se, &ce);
        float sig = (idx == KCOMP) ? 0.01f : 0.94f;
        float altx = fmaf(sig, re * ce, mu0x), alty = fmaf(sig, re * se, mu0y);

        bool ring = idx < KCOMP;
        f32x2 v; v.x = ring ? ringx : altx; v.y = ring ? ringy : alty;

        __builtin_nontemporal_store(v, (f32x2*)(out_xy + 2 * i));   // b64, TH=NT
        __builtin_nontemporal_store(idx, out_idx + i);              // b32, TH=NT
    }
}

// ---------------------------------------------------------------------------
extern "C" void kernel_launch(void* const* d_in, const int* in_sizes, int n_in,
                              void* d_out, int out_size, void* d_ws, size_t ws_size,
                              hipStream_t stream) {
    const float* c   = (const float*)d_in[0];
    const float* h   = (const float*)d_in[1];
    const float* W1  = (const float*)d_in[2];
    const float* b1  = (const float*)d_in[3];
    const float* W2  = (const float*)d_in[4];
    const float* b2  = (const float*)d_in[5];
    const float* W3  = (const float*)d_in[6];
    const float* b3  = (const float*)d_in[7];
    const float* Wmu = (const float*)d_in[8];
    const float* bmu = (const float*)d_in[9];
    const float* WR  = (const float*)d_in[10];
    const float* bR  = (const float*)d_in[11];
    (void)in_sizes; (void)n_in; (void)ws_size;

    const long long N = (long long)out_size / 3;   // out = xy(2N f32) ++ idx(N i32)
    float* P       = (float*)d_ws;
    float* out_xy  = (float*)d_out;
    int*   out_idx = (int*)((float*)d_out + 2 * N);

    mrp_mlp_kernel<<<1, 32, 0, stream>>>(c, h, W1, b1, W2, b2, W3, b3,
                                         Wmu, bmu, WR, bR, P);

    const int threads = 256, spt = 8;
    long long blocks = (N + (long long)threads * spt - 1) / ((long long)threads * spt);
    if (blocks < 1) blocks = 1;
    if (blocks > 65536) blocks = 65536;
    mrp_sample_kernel<<<(unsigned)blocks, threads, 0, stream>>>(P, out_xy, out_idx, N);
}
